// mtlAttention_57234734186967
// MI455X (gfx1250) — matrix operand order repaired and checked by reading, not compiled, not run on hardware
//
#include <hip/hip_runtime.h>
#include <hip/hip_bf16.h>

// ---------------------------------------------------------------------------
// Problem constants
// ---------------------------------------------------------------------------
#define BATCH 2048
#define MODM  128          // M modules
#define TWO_M 256          // 2*M

typedef __attribute__((ext_vector_type(8)))  float  v8f;
typedef __attribute__((ext_vector_type(16))) __bf16 v16bf;

// ---------------------------------------------------------------------------
// Kernel 1: fused projection GEMM for one task set.
//   H[b, j] = sum_k x[b,k] * w[j,k]   with w = concat(wx, wy) rows (j<128 -> wx)
// A operand = x rows (M dim = batch rows), B operand = weight rows (N dim).
// bf16 WMMA 16x16x32, f32 accumulate. Wave = 16 rows x 64 cols (4 C tiles).
// Block = 256 threads = 8 waves -> 32 rows x 256 cols. grid.x = 2048/32 = 64.
// grid.y = task (0..2), each with its own K.
// ---------------------------------------------------------------------------
struct ProjTask {
    const float* x;    // [B, K]
    const float* wx;   // [128, K]
    const float* wy;   // [128, K]
    float*       H;    // [B, 256] raw (un-normalized) projections
    int          K;
};

__device__ __forceinline__ void cvt8_to(v16bf& dst, int base, float4 a, float4 b) {
    dst[base + 0] = (__bf16)a.x; dst[base + 1] = (__bf16)a.y;
    dst[base + 2] = (__bf16)a.z; dst[base + 3] = (__bf16)a.w;
    dst[base + 4] = (__bf16)b.x; dst[base + 5] = (__bf16)b.y;
    dst[base + 6] = (__bf16)b.z; dst[base + 7] = (__bf16)b.w;
}

__global__ void __launch_bounds__(256)
proj_kernel(ProjTask t0, ProjTask t1, ProjTask t2) {
    ProjTask t = (blockIdx.y == 0) ? t0 : ((blockIdx.y == 1) ? t1 : t2);
    const int tid    = threadIdx.x;
    const int wave   = tid >> 5;
    const int lane   = tid & 31;
    const int hi     = lane >> 4;     // which K-half this lane holds
    const int lane16 = lane & 15;

    const int rowGroup = wave >> 2;               // 0..1
    const int colGroup = wave & 3;                // 0..3
    const int Rbase    = blockIdx.x * 32 + rowGroup * 16;
    const int Cbase    = colGroup * 64;
    const int K        = t.K;

    v8f acc[4] = {};   // four 16x16 f32 C tiles

    const int r = Rbase + lane16;                 // A row for this lane
    const float* xrow = t.x + (size_t)r * K;

    const float* wrow[4];
#pragma unroll
    for (int c = 0; c < 4; ++c) {
        int j = Cbase + c * 16 + lane16;          // output column = weight row
        wrow[c] = (j < MODM) ? (t.wx + (size_t)j * K)
                             : (t.wy + (size_t)(j - MODM) * K);
    }

    int kk = 0;
    for (; kk + 32 <= K; kk += 32) {
        // ----- A fragment: 16x32 bf16, contiguous k per lane half --------------------------
        // lane<16: K [kk..kk+16) in elems 0..15
        // lane>=16: shifted by +16
        v16bf a;
        {
            const float4* p0 = (const float4*)(xrow + kk + hi * 16);
            const float4* p1 = (const float4*)(xrow + kk + 8 + hi * 16);
            cvt8_to(a, 0, p0[0], p0[1]);
            cvt8_to(a, 8, p1[0], p1[1]);
        }
        // ----- 4 B fragments + WMMA ---------------------------------------
#pragma unroll
        for (int c = 0; c < 4; ++c) {
            v16bf b;
            const float4* q = (const float4*)(wrow[c] + kk + hi * 16);
            cvt8_to(b, 0, q[0], q[1]);
            cvt8_to(b, 8, q[2], q[3]);
            acc[c] = __builtin_amdgcn_wmma_f32_16x16x32_bf16(
                false, a, false, b, (short)0, acc[c], false, false);
        }
    }

    // ----- K tail (t3: 5000 = 156*32 + 8), guarded scalar loads -----------
    if (kk < K) {
        v16bf a;
#pragma unroll
        for (int e = 0; e < 16; ++e) {
            int k = kk + hi * 16 + e;
            a[e] = (k < K) ? (__bf16)xrow[k] : (__bf16)0.0f;
        }
#pragma unroll
        for (int c = 0; c < 4; ++c) {
            v16bf b;
#pragma unroll
            for (int e = 0; e < 16; ++e) {
                int k = kk + hi * 16 + e;
                b[e] = (k < K) ? (__bf16)wrow[c][k] : (__bf16)0.0f;
            }
            acc[c] = __builtin_amdgcn_wmma_f32_16x16x32_bf16(
                false, a, false, b, (short)0, acc[c], false, false);
        }
    }

    // ----- store C: lanes 0-15 -> M=v, N=lane; lanes 16-31 -> M=v+8 -------
#pragma unroll
    for (int c = 0; c < 4; ++c) {
        int col = Cbase + c * 16 + lane16;
#pragma unroll
        for (int v = 0; v < 8; ++v) {
            int row = Rbase + v + hi * 8;
            t.H[(size_t)row * TWO_M + col] = acc[c][v];
        }
    }
}

// ---------------------------------------------------------------------------
// Kernel 2: per-batch-element attention + towers. One block per b.
// ---------------------------------------------------------------------------
struct Tower {
    const float *w2, *b2, *w3, *b3, *w4, *b4, *w5, *b5;
};
struct AttnArgs {
    const float* H0;   // raw projections [B,256] per task
    const float* H1;
    const float* H2;
    Tower tw[3];
    float* out;        // [3*B]
};

#define APAD 129   // LDS stride for 128x128 matrix (conflict-free columns)

__global__ void __launch_bounds__(256)
attn_kernel(AttnArgs A) {
    extern __shared__ float lds[];
    float* Amat = lds;                       // 128*129
    float* vecs = Amat + 128 * APAD;         // 3 tasks * [x(128) | y(128)]
    float* accb = vecs + 3 * TWO_M;          // gc/mc/lc, 3 * 256
    float* h64  = accb + 3 * TWO_M;          // 3*64
    float* h32  = h64 + 192;                 // 3*32
    float* h16  = h32 + 96;                  // 3*16

    const int tid = threadIdx.x;
    const int b   = blockIdx.x;

    // ---- load raw projections, L2-normalize the (x,y) pair per module ----
    for (int idx = tid; idx < 3 * MODM; idx += 256) {
        int t = idx >> 7, m = idx & (MODM - 1);
        const float* H = ((t == 0) ? A.H0 : (t == 1) ? A.H1 : A.H2)
                         + (size_t)b * TWO_M;
        float hx = H[m], hy = H[MODM + m];
        float inv = rsqrtf(hx * hx + hy * hy);
        vecs[t * TWO_M + m]        = hx * inv;
        vecs[t * TWO_M + MODM + m] = hy * inv;
    }
    for (int idx = tid; idx < 3 * TWO_M; idx += 256) accb[idx] = 0.0f;
    __syncthreads();

    // ---- three attention pairs: (g,m) (g,l) (m,l) ------------------------
    for (int p = 0; p < 3; ++p) {
        const int ia = p >> 1;                 // 0,0,1
        const int ib = (p == 0) ? 1 : 2;       // 1,2,2
        const float* av = vecs + ia * TWO_M;
        const float* bv = vecs + ib * TWO_M;

        // energy E[m,n] = ax[m]*bx[n] + ay[m]*by[n]  (rank-2)
        for (int idx = tid; idx < MODM * MODM; idx += 256) {
            int m = idx >> 7, n = idx & (MODM - 1);
            Amat[m * APAD + n] = av[m] * bv[n] + av[MODM + m] * bv[MODM + n];
        }
        __syncthreads();

        // softmax over last dim, one thread per row
        if (tid < MODM) {
            float* row = Amat + tid * APAD;
            float mx = row[0];
            for (int n = 1; n < MODM; ++n) mx = fmaxf(mx, row[n]);
            float s = 0.0f;
            for (int n = 0; n < MODM; ++n) { float e = __expf(row[n] - mx); row[n] = e; s += e; }
            float inv = 1.0f / s;
            for (int n = 0; n < MODM; ++n) row[n] *= inv;
        }
        __syncthreads();

        // left(first, A)  -> accumulates into first operand's context
        // right(second, A)-> accumulates into second operand's context
        const int leftOut  = (p == 2) ? 1 : 0;   // gc, gc, mc
        const int rightOut = (p == 0) ? 1 : 2;   // mc, lc, lc
        const int k = tid >> 7, j = tid & (MODM - 1);

        const float* xlft = vecs + ia * TWO_M + k * MODM;
        float s = 0.0f;
        for (int m = 0; m < MODM; ++m) s += xlft[m] * Amat[m * APAD + j];
        accb[leftOut * TWO_M + k * MODM + j] += s;

        const float* xrgt = vecs + ib * TWO_M + k * MODM;
        float s2 = 0.0f;
        for (int n = 0; n < MODM; ++n) s2 += xrgt[n] * Amat[j * APAD + n];
        accb[rightOut * TWO_M + k * MODM + j] += s2;
        __syncthreads();
    }

    // ---- towers: 256 -> 64 -> 32 -> 16 -> 1, three tasks in parallel -----
    if (tid < 192) {
        int t = tid >> 6, o = tid & 63;
        const Tower tw = (t == 0) ? A.tw[0] : (t == 1) ? A.tw[1] : A.tw[2];
        const float* hin = accb + t * TWO_M;
        float s = tw.b2[o];
        for (int i = 0; i < 256; ++i) s += tw.w2[o * 256 + i] * hin[i];
        h64[t * 64 + o] = fmaxf(s, 0.0f);
    }
    __syncthreads();
    if (tid < 96) {
        int t = tid >> 5, o = tid & 31;
        const Tower tw = (t == 0) ? A.tw[0] : (t == 1) ? A.tw[1] : A.tw[2];
        float s = tw.b3[o];
        for (int i = 0; i < 64; ++i) s += tw.w3[o * 64 + i] * h64[t * 64 + i];
        h32[t * 32 + o] = fmaxf(s, 0.0f);
    }
    __syncthreads();
    if (tid < 48) {
        int t = tid >> 4, o = tid & 15;
        const Tower tw = (t == 0) ? A.tw[0] : (t == 1) ? A.tw[1] : A.tw[2];
        float s = tw.b4[o];
        for (int i = 0; i < 32; ++i) s += tw.w4[o * 32 + i] * h32[t * 32 + i];
        h16[t * 16 + o] = fmaxf(s, 0.0f);
    }
    __syncthreads();
    if (tid < 3) {
        int t = tid;
        const Tower tw = (t == 0) ? A.tw[0] : (t == 1) ? A.tw[1] : A.tw[2];
        float s = tw.b5[0];
        for (int i = 0; i < 16; ++i) s += tw.w5[i] * h16[t * 16 + i];
        A.out[t * BATCH + b] = 1.0f / (1.0f + __expf(-s));
    }
}

// ---------------------------------------------------------------------------
// Host side
// ---------------------------------------------------------------------------
extern "C" void kernel_launch(void* const* d_in, const int* in_sizes, int n_in,
                              void* d_out, int out_size, void* d_ws, size_t ws_size,
                              hipStream_t stream) {
    (void)in_sizes; (void)n_in; (void)out_size; (void)ws_size;

    auto P = [&](int i) { return (const float*)d_in[i]; };
    const float* xg = P(0);
    const float* xm = P(1);
    const float* xl = P(2);
    // params flat order per task: wx, wy, fc2_w, fc2_b, fc3_w, fc3_b,
    //                             fc4_w, fc4_b, fc5_w, fc5_b   (10 each)
    float* ws = (float*)d_ws;
    float* Hg = ws;
    float* Hm = ws + (size_t)BATCH * TWO_M;
    float* Hl = ws + 2 * (size_t)BATCH * TWO_M;

    ProjTask t0{xg, P(3),  P(4),  Hg, 20000};
    ProjTask t1{xm, P(13), P(14), Hm, 12000};
    ProjTask t2{xl, P(23), P(24), Hl, 5000};
    proj_kernel<<<dim3(BATCH / 32, 3), 256, 0, stream>>>(t0, t1, t2);

    AttnArgs A;
    A.H0 = Hg; A.H1 = Hm; A.H2 = Hl; A.out = (float*)d_out;
    for (int t = 0; t < 3; ++t) {
        int b0 = 3 + t * 10;
        A.tw[t] = Tower{P(b0 + 2), P(b0 + 3), P(b0 + 4), P(b0 + 5),
                        P(b0 + 6), P(b0 + 7), P(b0 + 8), P(b0 + 9)};
    }

    size_t shmem = (size_t)(128 * APAD + 3 * TWO_M + 3 * TWO_M + 192 + 96 + 48)
                   * sizeof(float);   // ~73 KB, fine on CDNA5 (320 KB/WGP)
    hipFuncSetAttribute(reinterpret_cast<const void*>(&attn_kernel),
                        hipFuncAttributeMaxDynamicSharedMemorySize, (int)shmem);
    attn_kernel<<<BATCH, 256, shmem, stream>>>(A);
}
